// GPT3SelfAttention_11330123727337
// MI455X (gfx1250) — compile-verified, matrix operand order checked by reading, not run on hardware
//
#include <hip/hip_runtime.h>
#include <hip/hip_bf16.h>

// ---------------------------------------------------------------------------
// GPT-3 self-attention block for MI455X (gfx1250, wave32, WMMA + TDM).
// B=2, S=2048, H=2048, NH=16, HD=128.
// Pipeline: f32->bf16 convert -> QKV GEMM (WMMA, TDM-staged A) ->
//           flash attention (WMMA) -> output projection GEMM (WMMA, fp32 out).
// ---------------------------------------------------------------------------

#define S_LEN 2048
#define HID   2048
#define NHEAD 16
#define HD    128
#define BSZ   2
#define MROWS (BSZ * S_LEN)   // 4096
#define N_QKV (3 * HID)       // 6144

#define USE_TDM 1

typedef __bf16 bf16;
typedef bf16  v16bf __attribute__((ext_vector_type(16)));
typedef float v8f   __attribute__((ext_vector_type(8)));
typedef unsigned int v4u __attribute__((ext_vector_type(4)));
typedef int v4i __attribute__((ext_vector_type(4)));
typedef int v8i __attribute__((ext_vector_type(8)));

#if __has_include(<hip/amd_detail/amd_gfx1250_TDM.h>)
#define TDM_6ARG 1
#endif

// ---------------------------------------------------------------------------
// WMMA helper: D = A(16x32 bf16) * B(32x16 bf16) + C(16x16 f32)
// ---------------------------------------------------------------------------
__device__ __forceinline__ v8f wmma_bf16(v16bf a, v16bf b, v8f c) {
    return __builtin_amdgcn_wmma_f32_16x16x32_bf16(
        /*neg_a=*/false, a, /*neg_b=*/false, b,
        /*c_mod=*/(short)0, c, /*reuse_a=*/false, /*reuse_b=*/false);
}

// Load one 16x32 bf16 A/B fragment given the per-lane row pointer.
// ISA layout (CDNA5 ISA):
//   lanes 0-15 : elems 0..7 = K[kbase+0..7],  elems 8..15 = K[kbase+16..23]
//   lanes 16-31: elems 0..7 = K[kbase+8..15], elems 8..15 = K[kbase+24..31]
__device__ __forceinline__ v16bf load_frag16(const bf16* rowPtr, int kbase, int half) {
    const int c0 = kbase + (half ? 8 : 0);
    union { uint4 u[2]; v16bf v; } t;
    t.u[0] = *reinterpret_cast<const uint4*>(rowPtr + c0);
    t.u[1] = *reinterpret_cast<const uint4*>(rowPtr + c0 + 16);
    return t.v;
}

// ---------------------------------------------------------------------------
// Tensor Data Mover: load a 2D bf16 tile (rows x 32) global -> LDS, with LDS
// padding of 4 dwords after every 16 dwords so each 32-elem row lands on a
// 40-elem (80B) stride — matching the padded LDS layout of the fragment
// loads.  D# packing per CDNA5 ISA.
// ---------------------------------------------------------------------------
#if USE_TDM
__device__ __forceinline__ void tdm_load_tile_rows_x32(
    const bf16* gsrc, bf16* ldst, unsigned rows, unsigned long long stride_elems)
{
    unsigned long long ga = (unsigned long long)(const void*)gsrc;
    unsigned lds = (unsigned)(unsigned long long)(void*)ldst;  // low 32b = LDS offset

    // ---- group 0: count / lds_addr / global_addr / type ----
    v4u g0;
    g0[0] = 1u;                                        // count=1 (valid descriptor)
    g0[1] = lds;                                       // lds_addr
    g0[2] = (unsigned)(ga & 0xffffffffu);              // global_addr[31:0]
    g0[3] = (unsigned)((ga >> 32) & 0x01ffffffu)       // global_addr[56:32]
          | (2u << 30);                                // type = 2 ("image")

    // ---- group 1: sizes / tile dims / strides / LDS padding ----
    // data_size=1 (2B), pad_enable, pad_interval=3 (16 dwords), pad_amount=3 (4 dwords)
    unsigned dw0 = (1u << 16) | (1u << 20) | (3u << 22) | (3u << 25);
    unsigned long long tensor_dim0 = stride_elems;     // row length (OOB guard)
    unsigned long long tensor_dim1 = 0x00400000ull;    // generous row count
    unsigned long long tile_dim0   = 32;               // contiguous K elems
    unsigned long long tile_dim1   = rows;
    unsigned long long tile_dim2   = 0;
    unsigned long long stride0     = stride_elems;     // elems between rows
    unsigned long long q0 = (unsigned long long)dw0 | (tensor_dim0 << 48);
    unsigned long long q1 = (tensor_dim0 >> 16) | (tensor_dim1 << 16) | (tile_dim0 << 48);
    unsigned long long q2 = tile_dim1 | (tile_dim2 << 16) | (stride0 << 32);
    unsigned long long q3 = (stride0 >> 32);           // stride0[47:32]; stride1=0
    v8i g1;
    g1[0] = (int)(unsigned)q0; g1[1] = (int)(unsigned)(q0 >> 32);
    g1[2] = (int)(unsigned)q1; g1[3] = (int)(unsigned)(q1 >> 32);
    g1[4] = (int)(unsigned)q2; g1[5] = (int)(unsigned)(q2 >> 32);
    g1[6] = (int)(unsigned)q3; g1[7] = (int)(unsigned)(q3 >> 32);

    v4i z4 = {0, 0, 0, 0};
#if defined(TDM_6ARG)
    v8i z8 = {0, 0, 0, 0, 0, 0, 0, 0};
    __builtin_amdgcn_tensor_load_to_lds(g0, g1, z4, z4, z8, 0);
#else
    __builtin_amdgcn_tensor_load_to_lds(g0, g1, z4, z4, 0);
#endif
}
#endif

// ---------------------------------------------------------------------------
// fp32 -> bf16 conversion (vectorized x4)
// ---------------------------------------------------------------------------
__global__ void __launch_bounds__(256)
cvt_f32_bf16_kernel(const float* __restrict__ in, bf16* __restrict__ out, int n4) {
    int i = blockIdx.x * 256 + threadIdx.x;
    if (i >= n4) return;
    float4 v = reinterpret_cast<const float4*>(in)[i];
    union { bf16 b[4]; uint2 u; } o;
    o.b[0] = (bf16)v.x; o.b[1] = (bf16)v.y;
    o.b[2] = (bf16)v.z; o.b[3] = (bf16)v.w;
    reinterpret_cast<uint2*>(out)[i] = o.u;
}

// ---------------------------------------------------------------------------
// Shared 128x128 GEMM tile engine (bf16 in, f32 accum).
// Block: 256 threads = 8 waves, wave tile 32x64 (2x4 WMMA tiles), BK = 32.
// A tile is DMA'd into LDS by the Tensor Data Mover (wave 0 issues, waits on
// TENSORcnt); B tile is transposed into n-major LDS by all threads.
// ---------------------------------------------------------------------------
__device__ __forceinline__ void gemm_tile_128x128(
    const bf16* __restrict__ A, const bf16* __restrict__ Bm,
    int K, int ldb, int m0, int n0, v8f acc[2][4])
{
    __shared__ bf16 As[128 * 40];   // [m][k], row stride 40 (80B = 5*16B)
    __shared__ bf16 Bs[128 * 40];   // [n][k], row stride 40

    const int t    = threadIdx.x;
    const int lane = t & 31;
    const int wv   = t >> 5;
    const int wm   = wv & 3;    // 0..3 -> M offset wm*32
    const int wn   = wv >> 2;   // 0..1 -> N offset wn*64
    const int lm   = lane & 15;
    const int half = lane >> 4;

    const v8f zero8 = {0, 0, 0, 0, 0, 0, 0, 0};
#pragma unroll
    for (int mt = 0; mt < 2; ++mt)
#pragma unroll
        for (int nt = 0; nt < 4; ++nt) acc[mt][nt] = zero8;

    for (int k0 = 0; k0 < K; k0 += 32) {
        // --- stage A: 128x32 bf16 tile via TDM (padded to stride 40) ---
#if USE_TDM
        if (wv == 0) {
            tdm_load_tile_rows_x32(A + (size_t)m0 * K + k0, As, 128,
                                   (unsigned long long)K);
        }
#else
#pragma unroll
        for (int it = 0; it < 2; ++it) {
            int idx = t + it * 256;
            int row = idx >> 2;
            int c8  = (idx & 3) << 3;
            uint4 v = *reinterpret_cast<const uint4*>(A + (size_t)(m0 + row) * K + k0 + c8);
            *reinterpret_cast<uint4*>(&As[row * 40 + c8]) = v;
        }
#endif
        // --- prefetch next K-step tiles into cache ---
        if (k0 + 32 < K) {
            __builtin_prefetch(A + (size_t)(m0 + (t >> 1)) * K + (k0 + 32) + (t & 1) * 16, 0, 1);
            __builtin_prefetch(Bm + (size_t)(k0 + 32 + (t >> 4)) * ldb + n0 + (t & 15) * 8, 0, 1);
        }
        // --- stage B with transpose: 32x128 -> Bs[n][k] ---
#pragma unroll
        for (int it = 0; it < 2; ++it) {
            int idx = t + it * 256;
            int kr  = idx >> 4;          // 0..31
            int n8  = (idx & 15) << 3;   // 0..120
            uint4 v = *reinterpret_cast<const uint4*>(Bm + (size_t)(k0 + kr) * ldb + n0 + n8);
            const bf16* e = reinterpret_cast<const bf16*>(&v);
#pragma unroll
            for (int j = 0; j < 8; ++j) Bs[(n8 + j) * 40 + kr] = e[j];
        }
#if USE_TDM
        if (wv == 0) __builtin_amdgcn_s_wait_tensorcnt(0);
#endif
        __syncthreads();

        v16bf af[2], bfr[4];
#pragma unroll
        for (int mt = 0; mt < 2; ++mt)
            af[mt] = load_frag16(&As[(wm * 32 + mt * 16 + lm) * 40], 0, half);
#pragma unroll
        for (int nt = 0; nt < 4; ++nt)
            bfr[nt] = load_frag16(&Bs[(wn * 64 + nt * 16 + lm) * 40], 0, half);
#pragma unroll
        for (int mt = 0; mt < 2; ++mt)
#pragma unroll
            for (int nt = 0; nt < 4; ++nt)
                acc[mt][nt] = wmma_bf16(af[mt], bfr[nt], acc[mt][nt]);
        __syncthreads();
    }
}

// ---------------------------------------------------------------------------
// QKV GEMM:  X[4096,2048] @ Wqkv[2048,6144] + bqkv
// Epilogue scatters:  Q,K -> [B,NH,S,HD] bf16,  V -> transposed [B,NH,HD,S].
// ---------------------------------------------------------------------------
__global__ void __launch_bounds__(256)
qkv_gemm_kernel(const bf16* __restrict__ X, const bf16* __restrict__ W,
                const float* __restrict__ bias,
                bf16* __restrict__ Qb, bf16* __restrict__ Kb, bf16* __restrict__ Vt)
{
    const int m0 = blockIdx.y * 128;
    const int n0 = blockIdx.x * 128;
    v8f acc[2][4];
    gemm_tile_128x128(X, W, HID, N_QKV, m0, n0, acc);

    const int lane = threadIdx.x & 31;
    const int wv   = threadIdx.x >> 5;
    const int wm   = wv & 3, wn = wv >> 2;
    const int lm   = lane & 15, half = lane >> 4;

#pragma unroll
    for (int mt = 0; mt < 2; ++mt)
#pragma unroll
        for (int nt = 0; nt < 4; ++nt)
#pragma unroll
            for (int r = 0; r < 8; ++r) {
                int gm = m0 + wm * 32 + mt * 16 + r + 8 * half;   // row in [0,4096)
                int gn = n0 + wn * 64 + nt * 16 + lm;             // col in [0,6144)
                float val = acc[mt][nt][r] + bias[gn];
                int b = gm >> 11;            // /S_LEN
                int s = gm & (S_LEN - 1);
                int part = gn / HID;         // 0=Q 1=K 2=V
                int c    = gn - part * HID;
                int head = c >> 7;           // /HD
                int hd   = c & (HD - 1);
                bf16 bv = (bf16)val;
                size_t bh = (size_t)(b * NHEAD + head);
                if (part == 0)      Qb[(bh * S_LEN + s) * HD + hd] = bv;
                else if (part == 1) Kb[(bh * S_LEN + s) * HD + hd] = bv;
                else                Vt[(bh * HD + hd) * S_LEN + s] = bv;
            }
}

// ---------------------------------------------------------------------------
// Flash attention.  Grid: (S/64, B*NH), block = 128 threads = 4 waves.
// Each wave owns 16 query rows; loops over keys 32 at a time.
// scores = (Q.K^T)/sqrt(HD), masked entries = -10000 (matches reference
// affine mask), online softmax, ctx = P.V  -> Ctx bf16 [B,S,H].
// ---------------------------------------------------------------------------
__global__ void __launch_bounds__(128)
attn_kernel(const bf16* __restrict__ Qb, const bf16* __restrict__ Kb,
            const bf16* __restrict__ Vt, bf16* __restrict__ Ctx)
{
    __shared__ bf16 P_lds[4][16][40];   // per-wave 16x32 prob tile (padded)

    const int lane = threadIdx.x & 31;
    const int wv   = threadIdx.x >> 5;
    const int lm   = lane & 15;
    const int half = lane >> 4;
    const int bh   = blockIdx.y;                  // b*NH + h
    const int q0w  = blockIdx.x * 64 + wv * 16;   // first query row of wave

    const bf16* Qh = Qb + (size_t)bh * S_LEN * HD;
    const bf16* Kh = Kb + (size_t)bh * S_LEN * HD;
    const bf16* Vh = Vt + (size_t)bh * HD * S_LEN;

    // Q fragments (16 rows x 128), kept in registers for the whole kernel.
    v16bf qf[4];
    {
        const bf16* qrow = Qh + (size_t)(q0w + lm) * HD;
#pragma unroll
        for (int kc = 0; kc < 4; ++kc) qf[kc] = load_frag16(qrow, kc * 32, half);
    }

    const v8f zero8 = {0, 0, 0, 0, 0, 0, 0, 0};
    v8f ctx[8];
#pragma unroll
    for (int nt = 0; nt < 8; ++nt) ctx[nt] = zero8;
    float rmax[8], rsum[8];
#pragma unroll
    for (int r = 0; r < 8; ++r) { rmax[r] = -1e30f; rsum[r] = 0.0f; }

    const float scale = 0.0883883476483184f;   // 1/sqrt(128)
    const int jmax = q0w + 15;                 // causal: max key index

    for (int j0 = 0; j0 <= jmax; j0 += 32) {
        // ---- scores for 32 keys: two 16x16 f32 tiles ----
        v8f s0 = zero8, s1 = zero8;
#pragma unroll
        for (int kc = 0; kc < 4; ++kc) {
            v16bf kf0 = load_frag16(Kh + (size_t)(j0 + lm) * HD,      kc * 32, half);
            v16bf kf1 = load_frag16(Kh + (size_t)(j0 + 16 + lm) * HD, kc * 32, half);
            s0 = wmma_bf16(qf[kc], kf0, s0);
            s1 = wmma_bf16(qf[kc], kf1, s1);
        }
        // ---- mask + online softmax (row = r + 8*half, col = lm / lm+16) ----
#pragma unroll
        for (int r = 0; r < 8; ++r) {
            int m = r + 8 * half;
            int q = q0w + m;
            float a  = (j0 + lm      <= q) ? s0[r] * scale : -10000.0f;
            float b2 = (j0 + 16 + lm <= q) ? s1[r] * scale : -10000.0f;
            float t = fmaxf(a, b2);
#pragma unroll
            for (int off = 8; off >= 1; off >>= 1) t = fmaxf(t, __shfl_xor(t, off, 32));
            float mnew  = fmaxf(rmax[r], t);
            float alpha = __expf(rmax[r] - mnew);
            rmax[r] = mnew;
            float p0 = __expf(a - mnew);
            float p1 = __expf(b2 - mnew);
            float ps = p0 + p1;
#pragma unroll
            for (int off = 8; off >= 1; off >>= 1) ps += __shfl_xor(ps, off, 32);
            rsum[r] = rsum[r] * alpha + ps;
            P_lds[wv][m][lm]      = (bf16)p0;
            P_lds[wv][m][16 + lm] = (bf16)p1;
#pragma unroll
            for (int nt = 0; nt < 8; ++nt) ctx[nt][r] *= alpha;
        }
        __builtin_amdgcn_wave_barrier();   // LDS is in-order per wave

        // ---- P (16x32, A-layout via LDS bounce) times V (32x128) ----
        v16bf pf = load_frag16(&P_lds[wv][lm][0], 0, half);
#pragma unroll
        for (int nt = 0; nt < 8; ++nt) {
            const bf16* vrow = Vh + (size_t)(nt * 16 + lm) * S_LEN + j0;
            v16bf vf = load_frag16(vrow, 0, half);
            ctx[nt] = wmma_bf16(pf, vf, ctx[nt]);
        }
    }

    // ---- normalize and store ctx as bf16 [B,S,H] ----
    const int b = bh >> 4;            // /NHEAD
    const int h = bh & (NHEAD - 1);
#pragma unroll
    for (int r = 0; r < 8; ++r) {
        float inv = 1.0f / rsum[r];
        int m = r + 8 * half;
        int q = q0w + m;
        size_t rowoff = ((size_t)(b * S_LEN + q)) * HID + h * HD;
#pragma unroll
        for (int nt = 0; nt < 8; ++nt)
            Ctx[rowoff + nt * 16 + lm] = (bf16)(ctx[nt][r] * inv);
    }
}

// ---------------------------------------------------------------------------
// Output projection:  Ctx[4096,2048] @ Wo[2048,2048] + bo -> fp32 out
// ---------------------------------------------------------------------------
__global__ void __launch_bounds__(256)
proj_gemm_kernel(const bf16* __restrict__ Ctx, const bf16* __restrict__ Wob,
                 const float* __restrict__ bo, float* __restrict__ out)
{
    const int m0 = blockIdx.y * 128;
    const int n0 = blockIdx.x * 128;
    v8f acc[2][4];
    gemm_tile_128x128(Ctx, Wob, HID, HID, m0, n0, acc);

    const int lane = threadIdx.x & 31;
    const int wv   = threadIdx.x >> 5;
    const int wm   = wv & 3, wn = wv >> 2;
    const int lm   = lane & 15, half = lane >> 4;

#pragma unroll
    for (int mt = 0; mt < 2; ++mt)
#pragma unroll
        for (int nt = 0; nt < 4; ++nt)
#pragma unroll
            for (int r = 0; r < 8; ++r) {
                int gm = m0 + wm * 32 + mt * 16 + r + 8 * half;
                int gn = n0 + wn * 64 + nt * 16 + lm;
                out[(size_t)gm * HID + gn] = acc[mt][nt][r] + bo[gn];
            }
}

// ---------------------------------------------------------------------------
// Launch
// ---------------------------------------------------------------------------
extern "C" void kernel_launch(void* const* d_in, const int* in_sizes, int n_in,
                              void* d_out, int out_size, void* d_ws, size_t ws_size,
                              hipStream_t stream)
{
    (void)in_sizes; (void)n_in; (void)out_size; (void)ws_size;

    const float* hidden = (const float*)d_in[0];
    // d_in[1] = ltor_mask: causal, reproduced analytically in attn_kernel.
    const float* Wqkv = (const float*)d_in[2];
    const float* bqkv = (const float*)d_in[3];
    const float* Wo   = (const float*)d_in[4];
    const float* bo   = (const float*)d_in[5];
    float* out = (float*)d_out;

    // Workspace carve (all region sizes are multiples of 256 bytes).
    char* ws = (char*)d_ws;
    size_t off = 0;
    auto carve = [&](size_t bytes) -> char* {
        char* p = ws + off;
        off += (bytes + 255) & ~(size_t)255;
        return p;
    };
    bf16* Xb  = (bf16*)carve((size_t)MROWS * HID   * sizeof(bf16));  // 16.8 MB
    bf16* Wqb = (bf16*)carve((size_t)HID   * N_QKV * sizeof(bf16));  // 25.2 MB
    bf16* Wob = (bf16*)carve((size_t)HID   * HID   * sizeof(bf16));  //  8.4 MB
    bf16* Qb  = (bf16*)carve((size_t)MROWS * HID   * sizeof(bf16));  // 16.8 MB
    bf16* Kb  = (bf16*)carve((size_t)MROWS * HID   * sizeof(bf16));  // 16.8 MB
    bf16* Vt  = (bf16*)carve((size_t)MROWS * HID   * sizeof(bf16));  // 16.8 MB
    bf16* Ctx = (bf16*)carve((size_t)MROWS * HID   * sizeof(bf16));  // 16.8 MB

    // 1) fp32 -> bf16 converts
    cvt_f32_bf16_kernel<<<(MROWS * HID / 4 + 255) / 256, 256, 0, stream>>>(hidden, Xb, MROWS * HID / 4);
    cvt_f32_bf16_kernel<<<(HID * N_QKV / 4 + 255) / 256, 256, 0, stream>>>(Wqkv, Wqb, HID * N_QKV / 4);
    cvt_f32_bf16_kernel<<<(HID * HID / 4 + 255) / 256, 256, 0, stream>>>(Wo, Wob, HID * HID / 4);

    // 2) fused QKV projection (WMMA GEMM + head-layout scatter)
    dim3 g1(N_QKV / 128, MROWS / 128);
    qkv_gemm_kernel<<<g1, 256, 0, stream>>>(Xb, Wqb, bqkv, Qb, Kb, Vt);

    // 3) causal flash attention (WMMA)
    dim3 g2(S_LEN / 64, BSZ * NHEAD);
    attn_kernel<<<g2, 128, 0, stream>>>(Qb, Kb, Vt, Ctx);

    // 4) output projection (WMMA GEMM, fp32 + bias)
    dim3 g3(HID / 128, MROWS / 128);
    proj_gemm_kernel<<<g3, 256, 0, stream>>>(Ctx, Wob, bo, out);
}
